// BinaryLinear_44057774522911
// MI455X (gfx1250) — matrix operand order repaired and checked by reading, not compiled, not run on hardware
//
#include <hip/hip_runtime.h>

typedef __attribute__((ext_vector_type(16))) _Float16     v16h;
typedef __attribute__((ext_vector_type(8)))  float        v8f;
typedef __attribute__((ext_vector_type(4)))  float        v4f;
typedef __attribute__((ext_vector_type(4)))  unsigned int v4u;

#define GLOBAL_AS __attribute__((address_space(1)))

namespace {
constexpr int Mdim = 8192;   // rows of x
constexpr int Ndim = 4096;   // OUT
constexpr int Kdim = 4096;   // IN

// ---- pass-2 (GEMM) tiling: 8 waves (4 in M x 2 in N), wave tile 64x64
constexpr int BM = 256, BN = 128;
constexpr int WM = 64,  WN = 64;
constexpr int MFR = WM / 16;        // 4 M-fragments per wave
constexpr int NFR = WN / 16;        // 4 N-fragments per wave
constexpr int KT  = 32;             // K per WMMA

// ---- LDS staging: KB halves of K double-buffered, padded stride.
// stride 72 halves = 36 dwords: 36*r mod 64 distinct for r=0..15 and
// 16-lane x 4-dword b128 reads tile all 64 banks -> conflict-free.
constexpr int KB         = 64;
constexpr int LDS_STRIDE = KB + 8;           // 72 halves = 144 B (16B aligned)
constexpr int A_TILE     = BM * LDS_STRIDE;  // halves (36 KB)
constexpr int B_TILE     = BN * LDS_STRIDE;  // halves (18 KB)
constexpr int BUF_HALVES = A_TILE + B_TILE;  // one buffer (54 KB)

union Frag  { v16h h; v4u q[2]; };           // 16 f16 = 2x b128
union Pack8 { _Float16 e[8]; v4u q; };       // 8 f16 = 1x b128
} // namespace

// Async DMA: global -> LDS, 16 bytes per lane, tracked by ASYNCcnt.
// ISA (08_async_tensor 4.4): INST_OFFSET is added to BOTH the LDS address
// and the global address, so one base pair serves a whole row segment.
template <int OFF>
__device__ __forceinline__ void async_b128_off(unsigned lds_byte_addr,
                                               const _Float16* gaddr) {
  asm volatile("global_load_async_to_lds_b128 %0, %1, off offset:%2"
               :
               : "v"(lds_byte_addr), "v"((unsigned long long)(size_t)gaddr),
                 "n"(OFF)
               : "memory");
}
__device__ __forceinline__ void wait_async0() {
  asm volatile("s_wait_asynccnt 0x0" ::: "memory");
}

// ---------------------------------------------------------------------------
// Pass 1a: x f32 -> f16 (RTE), 8 elements/thread, pure streaming.
// ---------------------------------------------------------------------------
__global__ __launch_bounds__(256) void cvt_x_f16_kernel(
    const float* __restrict__ Xp, _Float16* __restrict__ Xh)
{
  const size_t idx = ((size_t)blockIdx.x * 256 + threadIdx.x) * 8;
  v4f p0 = *(const GLOBAL_AS v4f*)(Xp + idx);
  v4f p1 = *(const GLOBAL_AS v4f*)(Xp + idx + 4);
  Pack8 o;
#pragma unroll
  for (int i = 0; i < 4; ++i) {
    o.e[i]     = (_Float16)p0[i];
    o.e[4 + i] = (_Float16)p1[i];
  }
  *(GLOBAL_AS v4u*)(Xh + idx) = o.q;
}

// ---------------------------------------------------------------------------
// Pass 1b: W f32 -> sign(W) in f16 (+1.0 / -1.0), 8 elements/thread.
// ---------------------------------------------------------------------------
__global__ __launch_bounds__(256) void bin_w_f16_kernel(
    const float* __restrict__ Wp, _Float16* __restrict__ Wh)
{
  const size_t idx = ((size_t)blockIdx.x * 256 + threadIdx.x) * 8;
  v4u r0 = *(const GLOBAL_AS v4u*)(Wp + idx);
  v4u r1 = *(const GLOBAL_AS v4u*)(Wp + idx + 4);
  v4u o;
#pragma unroll
  for (int j = 0; j < 2; ++j) {
    o[j]     = 0x3C003C00u ^ (((r0[2 * j] >> 16) & 0x8000u) | (r0[2 * j + 1] & 0x80000000u));
    o[2 + j] = 0x3C003C00u ^ (((r1[2 * j] >> 16) & 0x8000u) | (r1[2 * j + 1] & 0x80000000u));
  }
  *(GLOBAL_AS v4u*)(Wh + idx) = o;
}

// ---------------------------------------------------------------------------
// Stage one K-tile (KB halves) of A (BM rows) and B (BN rows) into LDS.
// 12 async b128 transfers per thread (8 for A, 4 for B), addresses via
// instruction-offset immediates: zero per-transfer VALU.
// ---------------------------------------------------------------------------
__device__ __forceinline__ void stage_tiles(
    const _Float16* __restrict__ Xh, const _Float16* __restrict__ Wh,
    unsigned lbase, int mBlock, int nBlock, int k0, int tid)
{
  // A: thread t owns row t (BM == 256 threads), KB halves = 8 x b128
  {
    const _Float16* ga = Xh + (size_t)(mBlock + tid) * Kdim + k0;
    const unsigned  la = lbase + (unsigned)(tid * LDS_STRIDE) * 2u;
    async_b128_off<0>(la, ga);
    async_b128_off<16>(la, ga);
    async_b128_off<32>(la, ga);
    async_b128_off<48>(la, ga);
    async_b128_off<64>(la, ga);
    async_b128_off<80>(la, ga);
    async_b128_off<96>(la, ga);
    async_b128_off<112>(la, ga);
  }
  // B: 2 threads per row (BN == 128), each stages half the row (64 B)
  {
    const int row = tid & (BN - 1);
    const int seg = tid >> 7;  // 0 or 1 -> halves 0..31 / 32..63
    const _Float16* gb = Wh + (size_t)(nBlock + row) * Kdim + k0 + seg * 32;
    const unsigned  lb = lbase +
        (unsigned)(A_TILE + row * LDS_STRIDE + seg * 32) * 2u;
    async_b128_off<0>(lb, gb);
    async_b128_off<16>(lb, gb);
    async_b128_off<32>(lb, gb);
    async_b128_off<48>(lb, gb);
  }
}

// ---------------------------------------------------------------------------
// Pass 2: C = Xh * sign(W)h^T + bias. f16 WMMA, f32 accumulate.
// Double-buffered async-to-LDS staging; ds_load_b128 fragment reads.
// ---------------------------------------------------------------------------
__global__ __launch_bounds__(256, 1) void binlin_wmma_f16_kernel(
    const _Float16* __restrict__ Xh,
    const _Float16* __restrict__ Wh,
    const float* __restrict__ Bias,
    float* __restrict__ Out)
{
  __shared__ __align__(16) _Float16 smem[2][BUF_HALVES];

  const int tid  = threadIdx.x;
  const int lane = tid & 31;
  const int wave = tid >> 5;
  const int r    = lane & 15;
  const int half = lane >> 4;

  const int mBlock = blockIdx.y * BM;
  const int nBlock = blockIdx.x * BN;
  const int wm = (wave & 3) * WM;    // wave's M offset in block tile
  const int wn = (wave >> 2) * WN;   // wave's N offset in block tile

  v8f acc[MFR][NFR];
#pragma unroll
  for (int mf = 0; mf < MFR; ++mf)
#pragma unroll
    for (int nf = 0; nf < NFR; ++nf) {
      v8f z = {};
      acc[mf][nf] = z;
    }

  const unsigned lb0 = (unsigned)(size_t)&smem[0][0];
  const unsigned lb1 = (unsigned)(size_t)&smem[1][0];

  // Prologue: stage first K-tile into buffer 0.
  stage_tiles(Xh, Wh, lb0, mBlock, nBlock, 0, tid);

  constexpr int NKT = Kdim / KB;  // 64 K-tiles
  for (int kt = 0; kt < NKT; ++kt) {
    wait_async0();     // my async copies for buffer kt&1 have landed
    __syncthreads();   // everyone's copies landed; prev reads finished

    if (kt + 1 < NKT)  // stage next tile into the other buffer
      stage_tiles(Xh, Wh, (kt & 1) ? lb0 : lb1,
                  mBlock, nBlock, (kt + 1) * KB, tid);

    const _Float16* bufA = &smem[kt & 1][0];
    const _Float16* bufB = &smem[kt & 1][A_TILE];

#pragma unroll
    for (int kk = 0; kk < KB / KT; ++kk) {
      // A fragment (16x32): lane r = row; lanes 0-15 K 0..15,
      // lanes 16-31 K 16..31 (contiguous)
      Frag a[MFR];
#pragma unroll
      for (int mf = 0; mf < MFR; ++mf) {
        const _Float16* p = bufA + (wm + mf * 16 + r) * LDS_STRIDE
                                 + kk * KT + half * 16;
        a[mf].q[0] = *(const v4u*)(p);
        a[mf].q[1] = *(const v4u*)(p + 8);
      }
      // B fragment (32x16): lane r = column; lanes 0-15 K 0..15,
      // lanes 16-31 K 16..31 (contiguous)
      Frag b[NFR];
#pragma unroll
      for (int nf = 0; nf < NFR; ++nf) {
        const _Float16* p = bufB + (wn + nf * 16 + r) * LDS_STRIDE
                                 + kk * KT + half * 16;
        b[nf].q[0] = *(const v4u*)(p);
        b[nf].q[1] = *(const v4u*)(p + 8);
      }
#pragma unroll
      for (int mf = 0; mf < MFR; ++mf)
#pragma unroll
        for (int nf = 0; nf < NFR; ++nf)
          acc[mf][nf] = __builtin_amdgcn_wmma_f32_16x16x32_f16(
              false, a[mf].h, false, b[nf].h,
              (short)0, acc[mf][nf], false, false);
    }
  }

  // Epilogue: C/D layout = VGPR j, lanes 0-15 -> M=j, lanes 16-31 -> M=8+j.
#pragma unroll
  for (int nf = 0; nf < NFR; ++nf) {
    const int col = nBlock + wn + nf * 16 + r;
    const float bv = *(const GLOBAL_AS float*)(Bias + col);
#pragma unroll
    for (int mf = 0; mf < MFR; ++mf) {
      float* cp = Out + (size_t)(mBlock + wm + mf * 16 + half * 8) * Ndim + col;
#pragma unroll
      for (int j = 0; j < 8; ++j)
        *(GLOBAL_AS float*)(cp + (size_t)j * Ndim) = acc[mf][nf][j] + bv;
    }
  }
}

// ---------------------------------------------------------------------------
// Fallback (workspace too small): fused kernel, converts fragments inline.
// ---------------------------------------------------------------------------
__global__ __launch_bounds__(256, 1) void binlin_wmma_fused_kernel(
    const float* __restrict__ Xp,
    const float* __restrict__ Wp,
    const float* __restrict__ Bias,
    float* __restrict__ Out)
{
  const int lane = threadIdx.x & 31;
  const int wave = threadIdx.x >> 5;
  const int r    = lane & 15;
  const int half = lane >> 4;

  const int m0 = blockIdx.y * 128 + (wave & 3) * 32;
  const int n0 = blockIdx.x * 128 + (wave >> 2) * 64;

  v8f acc[2][4];
#pragma unroll
  for (int mf = 0; mf < 2; ++mf)
#pragma unroll
    for (int nf = 0; nf < 4; ++nf) {
      v8f z = {};
      acc[mf][nf] = z;
    }

  const float* xrow[2];
#pragma unroll
  for (int mf = 0; mf < 2; ++mf)
    xrow[mf] = Xp + (size_t)(m0 + mf * 16 + r) * Kdim;
  const float* wrow[4];
#pragma unroll
  for (int nf = 0; nf < 4; ++nf)
    wrow[nf] = Wp + (size_t)(n0 + nf * 16 + r) * Kdim;

  union AFrag { v16h h; _Float16 e[16]; };
  union BFrag { v16h h; unsigned int u[8]; };

  for (int k0 = 0; k0 < Kdim; k0 += KT) {
    const int ka = k0 + half * 16;
    AFrag a[2];
#pragma unroll
    for (int mf = 0; mf < 2; ++mf) {
      v4f p0 = *(const GLOBAL_AS v4f*)(xrow[mf] + ka);
      v4f p1 = *(const GLOBAL_AS v4f*)(xrow[mf] + ka + 4);
      v4f p2 = *(const GLOBAL_AS v4f*)(xrow[mf] + ka + 8);
      v4f p3 = *(const GLOBAL_AS v4f*)(xrow[mf] + ka + 12);
#pragma unroll
      for (int i = 0; i < 4; ++i) {
        a[mf].e[i]      = (_Float16)p0[i];
        a[mf].e[4 + i]  = (_Float16)p1[i];
        a[mf].e[8 + i]  = (_Float16)p2[i];
        a[mf].e[12 + i] = (_Float16)p3[i];
      }
    }
    const int kb = k0 + half * 16;
    BFrag b[4];
#pragma unroll
    for (int nf = 0; nf < 4; ++nf) {
      v4u q0 = *(const GLOBAL_AS v4u*)(wrow[nf] + kb);
      v4u q1 = *(const GLOBAL_AS v4u*)(wrow[nf] + kb + 4);
      v4u q2 = *(const GLOBAL_AS v4u*)(wrow[nf] + kb + 8);
      v4u q3 = *(const GLOBAL_AS v4u*)(wrow[nf] + kb + 12);
      unsigned int s[16];
#pragma unroll
      for (int i = 0; i < 4; ++i) {
        s[i] = q0[i]; s[4 + i] = q1[i]; s[8 + i] = q2[i]; s[12 + i] = q3[i];
      }
#pragma unroll
      for (int i = 0; i < 8; ++i) {
        unsigned int sg = ((s[2 * i] >> 16) & 0x8000u) | (s[2 * i + 1] & 0x80000000u);
        b[nf].u[i] = 0x3C003C00u ^ sg;
      }
    }
#pragma unroll
    for (int mf = 0; mf < 2; ++mf)
#pragma unroll
      for (int nf = 0; nf < 4; ++nf)
        acc[mf][nf] = __builtin_amdgcn_wmma_f32_16x16x32_f16(
            false, a[mf].h, false, b[nf].h, (short)0, acc[mf][nf], false, false);
  }

#pragma unroll
  for (int nf = 0; nf < 4; ++nf) {
    const int col = n0 + nf * 16 + r;
    const float bv = Bias[col];
#pragma unroll
    for (int mf = 0; mf < 2; ++mf) {
      float* cp = Out + (size_t)(m0 + mf * 16 + half * 8) * Ndim + col;
#pragma unroll
      for (int j = 0; j < 8; ++j)
        *(GLOBAL_AS float*)(cp + (size_t)j * Ndim) = acc[mf][nf][j] + bv;
    }
  }
}

extern "C" void kernel_launch(void* const* d_in, const int* in_sizes, int n_in,
                              void* d_out, int out_size, void* d_ws, size_t ws_size,
                              hipStream_t stream) {
  const float* x    = (const float*)d_in[0];
  const float* w    = (const float*)d_in[1];
  const float* bias = (const float*)d_in[2];
  float* out        = (float*)d_out;

  const size_t xh_bytes = (size_t)Mdim * Kdim * sizeof(_Float16);  // 64 MiB
  const size_t wh_bytes = (size_t)Ndim * Kdim * sizeof(_Float16);  // 32 MiB

  if (ws_size >= xh_bytes + wh_bytes) {
    _Float16* xh = (_Float16*)d_ws;
    _Float16* wh = (_Float16*)((char*)d_ws + xh_bytes);

    cvt_x_f16_kernel<<<(Mdim * (size_t)Kdim) / (256 * 8), 256, 0, stream>>>(x, xh);
    bin_w_f16_kernel<<<(Ndim * (size_t)Kdim) / (256 * 8), 256, 0, stream>>>(w, wh);

    dim3 grid(Ndim / BN, Mdim / BM);  // (32, 32)
    binlin_wmma_f16_kernel<<<grid, dim3(256), 0, stream>>>(xh, wh, bias, out);
  } else {
    dim3 grid(Ndim / 128, Mdim / 128);  // (32, 64)
    binlin_wmma_fused_kernel<<<grid, dim3(256), 0, stream>>>(x, w, bias, out);
  }

  (void)in_sizes; (void)n_in; (void)out_size;
}
